// BiLSTM_CRF_19370302505390
// MI455X (gfx1250) — matrix operand order repaired and checked by reading, not compiled, not run on hardware
//
#include <hip/hip_runtime.h>
#include <hip/hip_bf16.h>
#include <math.h>

// ---------------------------------------------------------------------------
// BiLSTM-CRF for MI455X (gfx1250, wave32).
//   V=200000, E=128, H=128 (per dir), 4H=512, K=7, T=4096
// Stage 2 (input projections) uses v_wmma_f32_16x16x32_f16.
// Stage 3 (recurrence) keeps w_hh (f32, 256KB) resident in the 320KB WGP LDS.
// ---------------------------------------------------------------------------

#define T_LEN 4096
#define E_DIM 128
#define H_DIM 128
#define G_DIM 512      // 4*H
#define N_ALL 1024     // both directions' gates
#define K_TAG 7
#define START_TAG 5
#define STOP_TAG 6
#define NEG_INF (-10000.0f)

typedef __attribute__((ext_vector_type(16))) _Float16 v16h;
typedef __attribute__((ext_vector_type(8)))  float    v8f;

// ---------------- workspace layout (bytes, all 4KB-aligned) ----------------
#define WS_XF16   0u                          // T*E f16          = 1,048,576
#define WS_WCAT   1048576u                    // 1024*128 f16     =   262,144
#define WS_BCAT   1310720u                    // 1024 f32         =     4,096
#define WS_XG     1314816u                    // T*1024 f32       = 16,777,216
#define WS_HF     18092032u                   // T*128 f32        =  2,097,152
#define WS_HB     20189184u                   // T*128 f32        =  2,097,152
#define WS_FEATS  22286336u                   // T*7 f32 (pad)    =   114,688
#define WS_BP     22401024u                   // T*7 i32 (pad)    =   114,688
// total ~21.5 MB

__device__ __forceinline__ float sigmoidf_(float x) {
  return 1.0f / (1.0f + __expf(-x));
}

// ---------------------------------------------------------------------------
// Kernel 1: pack w_ih_f / w_ih_b into one f16 weight block [1024][128]
//           (row n = gate row; cols 0..127 = E) and concat biases.
// ---------------------------------------------------------------------------
__global__ void pack_weights_kernel(const float* __restrict__ w_ih_f,
                                    const float* __restrict__ w_ih_b,
                                    const float* __restrict__ b_f,
                                    const float* __restrict__ b_b,
                                    _Float16* __restrict__ w_cat,
                                    float* __restrict__ b_cat) {
  int idx = blockIdx.x * blockDim.x + threadIdx.x;   // over 1024*128
  if (idx < N_ALL * E_DIM) {
    int r = idx >> 7, u = idx & 127;
    float v = (r < G_DIM) ? w_ih_f[r * E_DIM + u]
                          : w_ih_b[(r - G_DIM) * E_DIM + u];
    w_cat[idx] = (_Float16)v;
  }
  if (idx < G_DIM)                 b_cat[idx] = b_f[idx];
  else if (idx < N_ALL)            b_cat[idx] = b_b[idx - G_DIM];
}

// ---------------------------------------------------------------------------
// Kernel 2: embedding gather + f16 convert: x_f16[t][e] = embed[sent[t]][e]
// ---------------------------------------------------------------------------
__global__ void gather_embed_kernel(const int* __restrict__ sentence,
                                    const float* __restrict__ embed,
                                    _Float16* __restrict__ x_f16) {
  int idx = blockIdx.x * blockDim.x + threadIdx.x;   // over T*E
  if (idx >= T_LEN * E_DIM) return;
  int t = idx >> 7, e = idx & 127;
  int tok = sentence[t];
  x_f16[idx] = (_Float16)embed[tok * E_DIM + e];
}

// ---------------------------------------------------------------------------
// Kernel 3: WMMA GEMM  xg[T][1024] = x[T][128] @ w_cat^T + b_cat
//   block = 128 threads (4 waves); grid = (T/16, 1024/64)
//   each wave owns one 16x16 C tile; K loop = 4 x (16x16x32 f16 WMMA)
//   A/B fragment packing follows CDNA5 ISA 16-bit layouts.
// ---------------------------------------------------------------------------
__global__ __launch_bounds__(128)
void xgemm_wmma_kernel(const _Float16* __restrict__ x_f16,
                       const _Float16* __restrict__ w_cat,
                       const float* __restrict__ b_cat,
                       float* __restrict__ xg) {
  __shared__ _Float16 Atile[16 * E_DIM];     //  4 KB
  __shared__ _Float16 Btile[64 * E_DIM];     // 16 KB

  const int tid  = threadIdx.x;
  const int lane = tid & 31;
  const int wave = tid >> 5;

  // stage A (16 rows of x) and B (64 gate rows) into LDS
  for (int i = tid; i < 16 * E_DIM; i += 128)
    Atile[i] = x_f16[(blockIdx.x * 16 + (i >> 7)) * E_DIM + (i & 127)];
  for (int i = tid; i < 64 * E_DIM; i += 128)
    Btile[i] = w_cat[(blockIdx.y * 64 + (i >> 7)) * E_DIM + (i & 127)];
  __syncthreads();

  const int m  = lane & 15;     // A row / B column / C column index
  const int hi = lane >> 4;     // half-wave selector

  v8f acc = {};
  #pragma unroll
  for (int kb = 0; kb < E_DIM; kb += 32) {
    v16h a, b;
    #pragma unroll
    for (int v = 0; v < 8; ++v) {
      #pragma unroll
      for (int h2 = 0; h2 < 2; ++h2) {
        // A 16x32 f16: V0-3 -> K 0..15 (lanes<16: K0-7, lanes>=16: K8-15),
        //              V4-7 -> K16..31 likewise.
        int ka = ((v >> 2) << 4) + (hi << 3) + ((v & 3) << 1) + h2;
        a[v * 2 + h2] = Atile[m * E_DIM + kb + ka];
        // B 32x16 f16: same K layout as A (lanes<16: K0-7,16-23; lanes>=16: K8-15,24-31), N = lane&15
        int kbv = ((v >> 2) << 4) + (hi << 3) + ((v & 3) << 1) + h2;
        b[v * 2 + h2] = Btile[(wave * 16 + m) * E_DIM + kb + kbv];
      }
    }
    acc = __builtin_amdgcn_wmma_f32_16x16x32_f16(
        /*neg_a=*/false, a, /*neg_b=*/false, b,
        /*c_mod=*/(short)0, acc, /*reuse_a=*/false, /*reuse_b=*/false);
  }

  // C 16x16 f32: lane n (0..15)/n+16; VGPR r -> M=r (lanes<16) or r+8
  const int row0 = blockIdx.x * 16;
  const int col  = blockIdx.y * 64 + wave * 16 + m;
  const float bias = b_cat[col];
  #pragma unroll
  for (int r = 0; r < 8; ++r) {
    int mm = r + (hi << 3);
    xg[(row0 + mm) * N_ALL + col] = acc[r] + bias;
  }
}

// ---------------------------------------------------------------------------
// Kernel 4: sequential LSTM recurrence, one workgroup per direction.
//   512 threads; w_hh (f32) resident in LDS with 129-float row stride
//   (bank = (tid+u) % 64 -> conflict-free within a wave32).
//   Cell state c[u] lives in a register of thread u (u < 128).
// ---------------------------------------------------------------------------
#define WHH_STRIDE 129
__global__ __launch_bounds__(512)
void lstm_recurrence_kernel(const float* __restrict__ w_hh_f,
                            const float* __restrict__ w_hh_b,
                            const float* __restrict__ xg,
                            float* __restrict__ hf,
                            float* __restrict__ hb) {
  extern __shared__ float smem[];
  float* w_lds = smem;                              // 512*129 floats
  float* h_lds = w_lds + G_DIM * WHH_STRIDE;        // 128
  float* gates = h_lds + H_DIM;                     // 512

  const int tid = threadIdx.x;
  const int dir = blockIdx.x;                       // 0 = fwd, 1 = bwd
  const float* __restrict__ w_hh = dir ? w_hh_b : w_hh_f;
  float* __restrict__ hout       = dir ? hb : hf;

  for (int i = tid; i < G_DIM * H_DIM; i += 512) {
    int r = i >> 7, u = i & 127;
    w_lds[r * WHH_STRIDE + u] = w_hh[i];
  }
  if (tid < H_DIM) h_lds[tid] = 0.0f;
  float c = 0.0f;
  __syncthreads();

  const int col = dir * G_DIM + tid;
  for (int step = 0; step < T_LEN; ++step) {
    const int t = dir ? (T_LEN - 1 - step) : step;
    float acc = xg[t * N_ALL + col];
    if (step + 1 < T_LEN) {   // prefetch next step's gate inputs (global_prefetch_b8)
      const int tn = dir ? (t - 1) : (t + 1);
      __builtin_prefetch(&xg[tn * N_ALL + col], 0, 1);
    }
    const float* wr = &w_lds[tid * WHH_STRIDE];
    #pragma unroll 16
    for (int u = 0; u < H_DIM; ++u)
      acc = fmaf(wr[u], h_lds[u], acc);
    gates[tid] = acc;
    __syncthreads();

    if (tid < H_DIM) {
      float gi = sigmoidf_(gates[tid]);
      float gf = sigmoidf_(gates[H_DIM + tid]);
      float gg = tanhf(gates[2 * H_DIM + tid]);
      float go = sigmoidf_(gates[3 * H_DIM + tid]);
      c = gf * c + gi * gg;
      float hv = go * tanhf(c);
      h_lds[tid] = hv;
      hout[t * H_DIM + tid] = hv;
    }
    __syncthreads();
  }
}

// ---------------------------------------------------------------------------
// Kernel 5: feats[t][k] = [hf[t] | hb[t]] . w_out[k] + b_out[k]
// ---------------------------------------------------------------------------
__global__ __launch_bounds__(256)
void feats_kernel(const float* __restrict__ hf,
                  const float* __restrict__ hb,
                  const float* __restrict__ w_out,
                  const float* __restrict__ b_out,
                  float* __restrict__ feats) {
  __shared__ float w_s[K_TAG * 2 * H_DIM];
  __shared__ float b_s[K_TAG];
  for (int i = threadIdx.x; i < K_TAG * 2 * H_DIM; i += 256) w_s[i] = w_out[i];
  if (threadIdx.x < K_TAG) b_s[threadIdx.x] = b_out[threadIdx.x];
  __syncthreads();

  int t = blockIdx.x * blockDim.x + threadIdx.x;
  if (t >= T_LEN) return;
  const float* hfr = &hf[t * H_DIM];
  const float* hbr = &hb[t * H_DIM];
  #pragma unroll
  for (int k = 0; k < K_TAG; ++k) {
    float acc = b_s[k];
    const float* wk = &w_s[k * 2 * H_DIM];
    #pragma unroll 8
    for (int u = 0; u < H_DIM; ++u) acc = fmaf(hfr[u], wk[u], acc);
    #pragma unroll 8
    for (int u = 0; u < H_DIM; ++u) acc = fmaf(hbr[u], wk[H_DIM + u], acc);
    feats[t * K_TAG + k] = acc;
  }
}

// ---------------------------------------------------------------------------
// Kernel 6: Viterbi decode (single wave32). Lane k < 7 owns next-tag k.
//   d_out[0] = score, d_out[1..T] = path (as float).
// ---------------------------------------------------------------------------
__global__ __launch_bounds__(32)
void viterbi_kernel(const float* __restrict__ feats,
                    const float* __restrict__ trans,
                    int* __restrict__ bp,
                    float* __restrict__ out,
                    int out_size) {
  __shared__ float fv[K_TAG];
  __shared__ float fv_new[K_TAG];
  __shared__ float tr[K_TAG * K_TAG];

  const int k = threadIdx.x;
  if (k < K_TAG * K_TAG) tr[k] = trans[k];
  if (k < K_TAG) fv[k] = (k == START_TAG) ? 0.0f : NEG_INF;
  __syncthreads();

  for (int t = 0; t < T_LEN; ++t) {
    if (k < K_TAG) {
      float best = fv[0] + tr[k * K_TAG + 0];
      int arg = 0;
      #pragma unroll
      for (int p = 1; p < K_TAG; ++p) {
        float v = fv[p] + tr[k * K_TAG + p];
        if (v > best) { best = v; arg = p; }
      }
      bp[t * K_TAG + k] = arg;
      fv_new[k] = best + feats[t * K_TAG + k];
    }
    __syncthreads();
    if (k < K_TAG) fv[k] = fv_new[k];
    __syncthreads();
  }

  if (k == 0) {
    float best = fv[0] + tr[STOP_TAG * K_TAG + 0];
    int btag = 0;
    #pragma unroll
    for (int p = 1; p < K_TAG; ++p) {
      float v = fv[p] + tr[STOP_TAG * K_TAG + p];
      if (v > best) { best = v; btag = p; }
    }
    if (out_size > 0) out[0] = best;                       // score
    int tag = btag;
    if (T_LEN < out_size) out[T_LEN] = (float)btag;        // path[T-1] = best
    for (int t = T_LEN - 1; t >= 1; --t) {                 // path[t-1] = bp chain
      int prev = bp[t * K_TAG + tag];
      if (t < out_size) out[t] = (float)prev;
      tag = prev;
    }
  }
}

// ---------------------------------------------------------------------------
extern "C" void kernel_launch(void* const* d_in, const int* in_sizes, int n_in,
                              void* d_out, int out_size, void* d_ws, size_t ws_size,
                              hipStream_t stream) {
  const int*   sentence = (const int*)  d_in[0];
  const float* embed    = (const float*)d_in[1];
  const float* w_ih_f   = (const float*)d_in[2];
  const float* w_hh_f   = (const float*)d_in[3];
  const float* b_f      = (const float*)d_in[4];
  const float* w_ih_b   = (const float*)d_in[5];
  const float* w_hh_b   = (const float*)d_in[6];
  const float* b_b      = (const float*)d_in[7];
  const float* w_out    = (const float*)d_in[8];
  const float* b_out    = (const float*)d_in[9];
  const float* trans    = (const float*)d_in[10];

  char* ws = (char*)d_ws;
  _Float16* x_f16 = (_Float16*)(ws + WS_XF16);
  _Float16* w_cat = (_Float16*)(ws + WS_WCAT);
  float*    b_cat = (float*)   (ws + WS_BCAT);
  float*    xg    = (float*)   (ws + WS_XG);
  float*    hf    = (float*)   (ws + WS_HF);
  float*    hb    = (float*)   (ws + WS_HB);
  float*    feats = (float*)   (ws + WS_FEATS);
  int*      bp    = (int*)     (ws + WS_BP);

  // 1) pack input-projection weights to f16
  pack_weights_kernel<<<(N_ALL * E_DIM + 255) / 256, 256, 0, stream>>>(
      w_ih_f, w_ih_b, b_f, b_b, w_cat, b_cat);

  // 2) embedding gather -> f16
  gather_embed_kernel<<<(T_LEN * E_DIM + 255) / 256, 256, 0, stream>>>(
      sentence, embed, x_f16);

  // 3) xg = x @ [w_ih_f; w_ih_b]^T + b   (WMMA f16->f32)
  dim3 ggrid(T_LEN / 16, N_ALL / 64);
  xgemm_wmma_kernel<<<ggrid, 128, 0, stream>>>(x_f16, w_cat, b_cat, xg);

  // 4) bidirectional recurrence; w_hh resident in LDS (320KB WGP LDS)
  size_t lstm_smem = (size_t)(G_DIM * WHH_STRIDE + H_DIM + G_DIM) * sizeof(float);
  lstm_recurrence_kernel<<<2, 512, lstm_smem, stream>>>(w_hh_f, w_hh_b, xg, hf, hb);

  // 5) output projection to K=7 tag scores
  feats_kernel<<<T_LEN / 256, 256, 0, stream>>>(hf, hb, w_out, b_out, feats);

  // 6) Viterbi decode + backtrack
  viterbi_kernel<<<1, 32, 0, stream>>>(feats, trans, bp, (float*)d_out, out_size);
}
